// Model_3839700763130
// MI455X (gfx1250) — compile-verified, matrix operand order checked by reading, not run on hardware
//
#include <hip/hip_runtime.h>
#include <hip/hip_bf16.h>
#include <math.h>

// ---------------------------------------------------------------------------
// Reformer (LSH attention) forward for MI455X / gfx1250.
// wave32, v_wmma_f32_16x16x32_f16 for GEMMs AND chunk attention,
// TDM (tensor_load_to_lds) staging with double-buffered LDS.
// ---------------------------------------------------------------------------

typedef __attribute__((ext_vector_type(16))) _Float16 v16h;
typedef __attribute__((ext_vector_type(8)))  float    v8f;
typedef __attribute__((ext_vector_type(4)))  unsigned int v4u;
typedef __attribute__((ext_vector_type(8)))  int      v8i;
typedef __attribute__((ext_vector_type(4)))  int      v4i;

#define BDIM     16
#define T_LEN    1024
#define DMODEL   512
#define HEADS    8
#define DHEAD    64
#define NROUNDS  4
#define NBUCK    32
#define BH_TOT   128
#define NCHUNK   128
#define DFF      2048
#define PRED     256
#define COUT     7

#define BM 128
#define BN 64
#define BK 32

// -------- TDM availability (device pass only; host pass takes fallback) -----
#if defined(__HIP_DEVICE_COMPILE__) && __has_builtin(__builtin_amdgcn_tensor_load_to_lds)
#define USE_TDM 1
#else
#define USE_TDM 0
#endif

#if USE_TDM
// Build a 2D-tile D# (groups 0/1; higher groups zero) and issue
// TENSOR_LOAD_TO_LDS (6-arg form: g0,g1,g2,g3,g4,cpol).
// data_size = 2 bytes (f16). Tile is tile_cols x tile_rows, row stride = K.
__device__ inline void tdm_load_tile_f16(unsigned lds_byte_off,
                                         const _Float16* gsrc,
                                         int tile_cols, int tile_rows,
                                         int row_stride, int tensor_rows) {
  unsigned long long ga = (unsigned long long)(uintptr_t)gsrc;
  v4u g0;
  g0[0] = 1u;                                                  // count=1 (valid), user mode
  g0[1] = lds_byte_off;                                        // lds_addr (bytes)
  g0[2] = (unsigned)(ga & 0xffffffffull);                      // global_addr[31:0]
  g0[3] = (unsigned)((ga >> 32) & 0x1ffffffull) | (2u << 30);  // [56:32] | type=2
  unsigned td0 = (unsigned)row_stride;                         // tensor_dim0
  unsigned td1 = (unsigned)tensor_rows;                        // tensor_dim1
  unsigned long long st0 = (unsigned long long)row_stride;     // dim0 stride (elems)
  v8i g1;
  g1[0] = (int)(1u << 16);                                     // data_size=1 -> 2 bytes
  g1[1] = (int)(td0 << 16);                                    // tensor_dim0[15:0]
  g1[2] = (int)((td0 >> 16) | (td1 << 16));                    // dim0 hi | dim1 lo
  g1[3] = (int)((td1 >> 16) | ((unsigned)tile_cols << 16));    // dim1 hi | tile_dim0
  g1[4] = (int)(unsigned)tile_rows;                            // tile_dim1 (tile_dim2=0)
  g1[5] = (int)(unsigned)(st0 & 0xffffffffull);                // dim0_stride lo
  g1[6] = (int)(unsigned)((st0 >> 32) & 0xffffull);            // dim0_stride hi
  g1[7] = 0;
  v4i gz4; gz4[0] = 0; gz4[1] = 0; gz4[2] = 0; gz4[3] = 0;
  v8i gz8;
  for (int i = 0; i < 8; ++i) gz8[i] = 0;
  __builtin_amdgcn_tensor_load_to_lds(g0, g1, gz4, gz4, gz8, 0);
}

__device__ inline void tdm_wait0() {
#if __has_builtin(__builtin_amdgcn_s_wait_tensorcnt)
  __builtin_amdgcn_s_wait_tensorcnt(0);
#else
  asm volatile("s_wait_tensorcnt 0x0" ::: "memory");
#endif
}
#endif  // USE_TDM

__device__ inline v16h ld_frag16(const _Float16* lo, const _Float16* hi) {
  union { v16h v; float4 f[2]; } u;
  u.f[0] = *reinterpret_cast<const float4*>(lo);
  u.f[1] = *reinterpret_cast<const float4*>(hi);
  return u.v;
}

// ---------------------------- WMMA GEMM ------------------------------------
// C[M,N] = A[M,K](f16) @ Wt[N,K](f16, pre-transposed) (+bias)(+gelu).
// Block tile 128x64, 8 waves (2x4), wave tile 64x16, K step 32.
// Double-buffered LDS (24 KB dynamic), staged by TDM when available.
// OUT_MODE: 0 = f32 row-major, 1 = f16 row-major, 2 = f32 head-split.
template <int OUT_MODE, int ACT>
__global__ void __launch_bounds__(256)
gemm_wmma(const _Float16* __restrict__ A, const _Float16* __restrict__ Wt,
          const float* __restrict__ bias, void* __restrict__ Cptr,
          int M, int N, int K) {
  extern __shared__ float4 smem4[];
  _Float16* smem = (_Float16*)smem4;
  // LDS layout (bytes): sA[0] @0 (8K), sA[1] @8192, sB[0] @16384 (4K), sB[1] @20480

  const int tid  = threadIdx.x;
  const int lane = tid & 31;
  const int wave = tid >> 5;
  const int wm   = wave >> 2;
  const int wn   = wave & 3;
  const int bm0  = blockIdx.x * BM;
  const int bn0  = blockIdx.y * BN;

  v8f acc[4];
  for (int i = 0; i < 4; ++i)
    for (int j = 0; j < 8; ++j) acc[i][j] = 0.f;

  auto stage = [&](int buf, int kk) {
#if USE_TDM
    if (tid < 32) {  // one wave issues the DMA (TENSORcnt is per-wave)
      tdm_load_tile_f16((unsigned)(buf * (BM * BK * 2)),
                        A + (size_t)bm0 * K + kk, BK, BM, K, M - bm0);
      tdm_load_tile_f16((unsigned)(2 * (BM * BK * 2) + buf * (BN * BK * 2)),
                        Wt + (size_t)bn0 * K + kk, BK, BN, K, N - bn0);
    }
#else
    _Float16* dA = smem + buf * (BM * BK);
    _Float16* dB = smem + 2 * (BM * BK) + buf * (BN * BK);
    for (int vi = tid; vi < (BM * BK) / 8; vi += 256) {
      int r = vi >> 2, cv = (vi & 3) * 8;
      *(uint4*)(dA + r * BK + cv) =
          *(const uint4*)(A + (size_t)(bm0 + r) * K + kk + cv);
    }
    for (int vi = tid; vi < (BN * BK) / 8; vi += 256) {
      int n = vi >> 2, cv = (vi & 3) * 8;
      *(uint4*)(dB + n * BK + cv) =
          *(const uint4*)(Wt + (size_t)(bn0 + n) * K + kk + cv);
    }
#endif
  };

  stage(0, 0);
#if USE_TDM
  if (tid < 32) tdm_wait0();
#endif
  __syncthreads();

  for (int k0 = 0; k0 < K; k0 += BK) {
    const int buf = (k0 >> 5) & 1;
    if (k0 + BK < K) stage(buf ^ 1, k0 + BK);  // overlap next-tile DMA with WMMA

    const _Float16* pA = smem + buf * (BM * BK);
    const _Float16* pB = smem + 2 * (BM * BK) + buf * (BN * BK);
    const int kb = (lane < 16) ? 0 : 8;
    const int fn = wn * 16 + (lane & 15);
    v16h bf = ld_frag16(pB + fn * BK + kb, pB + fn * BK + kb + 16);
#pragma unroll
    for (int mt = 0; mt < 4; ++mt) {
      const int am = wm * 64 + mt * 16 + (lane & 15);
      v16h af = ld_frag16(pA + am * BK + kb, pA + am * BK + kb + 16);
      acc[mt] = __builtin_amdgcn_wmma_f32_16x16x32_f16(
          false, af, false, bf, (short)0, acc[mt], false, false);
    }
#if USE_TDM
    if (k0 + BK < K && tid < 32) tdm_wait0();
#endif
    __syncthreads();
  }

  const int colBase = bn0 + wn * 16 + (lane & 15);
  const int rowHalf = (lane >> 4) * 8;
#pragma unroll
  for (int mt = 0; mt < 4; ++mt) {
#pragma unroll
    for (int r = 0; r < 8; ++r) {
      int row = bm0 + wm * 64 + mt * 16 + rowHalf + r;
      float val = acc[mt][r];
      if (bias) val += bias[colBase];
      if (ACT == 1) val = 0.5f * val * (1.0f + erff(val * 0.70710678118654752f));
      if (OUT_MODE == 0) {
        ((float*)Cptr)[(size_t)row * N + colBase] = val;
      } else if (OUT_MODE == 1) {
        ((_Float16*)Cptr)[(size_t)row * N + colBase] = (_Float16)val;
      } else {  // head-split [b, h, t, dh]
        int b = row >> 10, t = row & 1023;
        int h = colBase >> 6, dh = colBase & 63;
        ((float*)Cptr)[((((size_t)b * HEADS + h) << 10) + t) * DHEAD + dh] = val;
      }
    }
  }
}

// ---------------------------- embedding ------------------------------------
__global__ void __launch_bounds__(256)
embed_kernel(const float* __restrict__ xe, const float* __restrict__ xme,
             const float* __restrict__ xd, const float* __restrict__ xmd,
             const float* __restrict__ convw, const float* __restrict__ markw,
             float* __restrict__ h, _Float16* __restrict__ h16) {
  size_t idx = (size_t)blockIdx.x * 256 + threadIdx.x;
  int d = idx & 511;
  int t = (int)((idx >> 9) & 1023);
  int b = (int)(idx >> 19);
  float acc = 0.f;
#pragma unroll
  for (int k = 0; k < 3; ++k) {
    int tt = (t + k - 1 + T_LEN) & (T_LEN - 1);
    const float* xr = (tt < 768) ? (xe + ((size_t)b * 768 + tt) * 7)
                                 : (xd + ((size_t)b * 512 + (tt - 512)) * 7);
#pragma unroll
    for (int i = 0; i < 7; ++i) acc += xr[i] * convw[((size_t)d * 7 + i) * 3 + k];
  }
  const float* mr = (t < 768) ? (xme + ((size_t)b * 768 + t) * 4)
                              : (xmd + ((size_t)b * 512 + (t - 512)) * 4);
#pragma unroll
  for (int m = 0; m < 4; ++m) acc += mr[m] * markw[(size_t)m * DMODEL + d];
  int pair = d >> 1;
  float div = expf((float)(2 * pair) * (-9.210340371976184f / 512.f));
  float ang = (float)t * div;
  acc += (d & 1) ? cosf(ang) : sinf(ang);
  h[idx] = acc;
  h16[idx] = (_Float16)acc;
}

// ------------------- LSH bucketing + stable counting sort -------------------
__global__ void __launch_bounds__(256)
lsh_bucket_sort(const float* __restrict__ qk, const float* __restrict__ rot,
                int* __restrict__ stb) {
  __shared__ float srot[DHEAD * 16];
  __shared__ int bkt[T_LEN];
  __shared__ int hist[NBUCK];
  __shared__ int boff[NBUCK];
  const int bh = blockIdx.x, nh = blockIdx.y, tid = threadIdx.x;

  for (int idx = tid; idx < DHEAD * 16; idx += 256) {
    int e = idx >> 4, i = idx & 15;
    srot[idx] = rot[((size_t)e * NROUNDS + nh) * 16 + i];
  }
  if (tid < NBUCK) hist[tid] = 0;
  __syncthreads();

  for (int t = tid; t < T_LEN; t += 256) {
    const float* q = qk + ((size_t)bh * T_LEN + t) * DHEAD;
    float rv[16];
#pragma unroll
    for (int i = 0; i < 16; ++i) rv[i] = 0.f;
    for (int e = 0; e < DHEAD; ++e) {
      float qe = q[e];
      const float* rr = srot + e * 16;
#pragma unroll
      for (int i = 0; i < 16; ++i) rv[i] += qe * rr[i];
    }
    float best = -3.4e38f; int bi = 0;
#pragma unroll
    for (int i = 0; i < 16; ++i) if (rv[i] > best) { best = rv[i]; bi = i; }
#pragma unroll
    for (int i = 0; i < 16; ++i) if (-rv[i] > best) { best = -rv[i]; bi = 16 + i; }
    bkt[t] = bi;
    atomicAdd(&hist[bi], 1);
  }
  __syncthreads();
  if (tid == 0) {
    int a = 0;
    for (int i = 0; i < NBUCK; ++i) { boff[i] = a; a += hist[i]; }
  }
  __syncthreads();
  for (int t = tid; t < T_LEN; t += 256) {
    int b = bkt[t];
    int r = boff[b];
    for (int t2 = 0; t2 < t; ++t2) r += (bkt[t2] == b) ? 1 : 0;
    stb[(size_t)bh * (NROUNDS * T_LEN) + nh * T_LEN + r] = t;
  }
}

// ------------------- chunked LSH attention (WMMA) ---------------------------
// One block (8 waves) per (bh, chunk). dots = bq.bk^T and bo = P.bv both run
// on v_wmma_f32_16x16x32_f16: 8 16x16 tiles per product, one tile per wave,
// 2 K-steps each. Scale/self-mask applied when spilling accumulators to LDS;
// logsumexp + exp stay scalar (2048 elements).
__global__ void __launch_bounds__(256)
lsh_attn(const float* __restrict__ qk, const float* __restrict__ v,
         const int* __restrict__ stb, _Float16* __restrict__ or16,
         float* __restrict__ lsebuf) {
  __shared__ __align__(16) _Float16 sqh[32][DHEAD];  // bq (f16)          4 KB
  __shared__ __align__(16) _Float16 skh[64][DHEAD];  // bk rows (f16)     8 KB
  __shared__ __align__(16) _Float16 svT[DHEAD][64];  // v transposed      8 KB
  __shared__ __align__(16) _Float16 pt[32][64];      // P (f16)           4 KB
  __shared__ float dt[32][64];                       //                   8 KB
  __shared__ float slse[32];
  __shared__ int qt[32];
  __shared__ int kt[64];
  const int bh = blockIdx.x, c = blockIdx.y, tid = threadIdx.x;
  const int lane = tid & 31, wave = tid >> 5;
  const int mt = wave >> 2;           // 0..1 (query row tile)
  const int nt = wave & 3;            // 0..3 (kv / dh column tile)
  const int* st = stb + (size_t)bh * (NROUNDS * T_LEN);
  const int nh = c >> 5;

  if (tid < 64) {
    int j = tid;
    int pc = (j < 32) ? c : ((c + NCHUNK - 1) & (NCHUNK - 1));
    int t = st[pc * 32 + (j & 31)];
    kt[j] = t;
    const float* qr = qk + ((size_t)bh * T_LEN + t) * DHEAD;
    float nrm = 0.f;
    for (int e = 0; e < DHEAD; ++e) nrm += qr[e] * qr[e];
    float inv = 1.f / fmaxf(sqrtf(nrm), 1e-12f);
    const float* vr = v + ((size_t)bh * T_LEN + t) * DHEAD;
    for (int e = 0; e < DHEAD; ++e) {
      skh[j][e] = (_Float16)(qr[e] * inv);
      svT[e][j] = (_Float16)vr[e];
    }
  } else if (tid < 96) {
    int i = tid - 64;
    int t = st[c * 32 + i];
    qt[i] = t;
    const float* qr = qk + ((size_t)bh * T_LEN + t) * DHEAD;
    for (int e = 0; e < DHEAD; ++e) sqh[i][e] = (_Float16)qr[e];
  }
  __syncthreads();

  const int kb0 = (lane < 16) ? 0 : 8;
  const int fr  = lane & 15;

  // ---- dots = bq . bk^T  (tile mt x nt, K = DHEAD in 2 steps) ----
  v8f acc;
  for (int j = 0; j < 8; ++j) acc[j] = 0.f;
#pragma unroll
  for (int ks = 0; ks < DHEAD; ks += 32) {
    v16h af = ld_frag16(&sqh[mt * 16 + fr][ks + kb0], &sqh[mt * 16 + fr][ks + kb0 + 16]);
    v16h bf = ld_frag16(&skh[nt * 16 + fr][ks + kb0], &skh[nt * 16 + fr][ks + kb0 + 16]);
    acc = __builtin_amdgcn_wmma_f32_16x16x32_f16(
        false, af, false, bf, (short)0, acc, false, false);
  }
  {
    const int col = nt * 16 + fr;
    const int rbase = mt * 16 + (lane >> 4) * 8;
#pragma unroll
    for (int r = 0; r < 8; ++r) {
      int row = rbase + r;
      float s = acc[r] * 0.125f;                 // dh^-0.5
      if (qt[row] == kt[col]) s = -5e4f;         // MASK_SELF
      dt[row][col] = s;
    }
  }
  __syncthreads();

  if (tid < 32) {
    int i = tid;
    float m = -3.4e38f;
    for (int j = 0; j < 64; ++j) m = fmaxf(m, dt[i][j]);
    float s = 0.f;
    for (int j = 0; j < 64; ++j) s += expf(dt[i][j] - m);
    float l = m + logf(s);
    slse[i] = l;
    lsebuf[(size_t)bh * (NROUNDS * T_LEN) + nh * T_LEN + qt[i]] = l;
  }
  __syncthreads();

  for (int idx = tid; idx < 2048; idx += 256) {
    int i = idx >> 6, j = idx & 63;
    pt[i][j] = (_Float16)expf(dt[i][j] - slse[i]);
  }
  __syncthreads();

  // ---- bo = P . bv  (tile mt x nt over dh, K = 64 kv in 2 steps) ----
  v8f acc2;
  for (int j = 0; j < 8; ++j) acc2[j] = 0.f;
#pragma unroll
  for (int ks = 0; ks < 64; ks += 32) {
    v16h af = ld_frag16(&pt[mt * 16 + fr][ks + kb0], &pt[mt * 16 + fr][ks + kb0 + 16]);
    v16h bf = ld_frag16(&svT[nt * 16 + fr][ks + kb0], &svT[nt * 16 + fr][ks + kb0 + 16]);
    acc2 = __builtin_amdgcn_wmma_f32_16x16x32_f16(
        false, af, false, bf, (short)0, acc2, false, false);
  }
  {
    const int e = nt * 16 + fr;
    const int rbase = mt * 16 + (lane >> 4) * 8;
#pragma unroll
    for (int r = 0; r < 8; ++r) {
      int i = rbase + r;
      or16[((((size_t)bh * NROUNDS + nh) << 10) + qt[i]) * DHEAD + e] = (_Float16)acc2[r];
    }
  }
}

// -------------------- combine rounds (softmax over lse) ---------------------
__global__ void __launch_bounds__(256)
lsh_combine(const _Float16* __restrict__ or16, const float* __restrict__ lsebuf,
            _Float16* __restrict__ merged) {
  size_t idx = (size_t)blockIdx.x * 256 + threadIdx.x;
  int e = (int)(idx & 63);
  int t = (int)((idx >> 6) & 1023);
  int bh = (int)(idx >> 16);
  float l[NROUNDS], m = -3.4e38f;
#pragma unroll
  for (int nh = 0; nh < NROUNDS; ++nh) {
    l[nh] = lsebuf[(size_t)bh * (NROUNDS * T_LEN) + nh * T_LEN + t];
    m = fmaxf(m, l[nh]);
  }
  float s = 0.f, o = 0.f;
#pragma unroll
  for (int nh = 0; nh < NROUNDS; ++nh) {
    float w = expf(l[nh] - m);
    s += w;
    o += w * (float)or16[((((size_t)bh * NROUNDS + nh) << 10) + t) * DHEAD + e];
  }
  o /= s;
  int b = bh >> 3, h = bh & 7;
  merged[((size_t)b * T_LEN + t) * DMODEL + h * DHEAD + e] = (_Float16)o;
}

// ------------- residual + layernorm (row = 512, f32 + f16 out) --------------
__global__ void __launch_bounds__(256)
add_ln(const float* __restrict__ x, const float* __restrict__ a,
       float* __restrict__ out, _Float16* __restrict__ out16,
       const float* __restrict__ g, const float* __restrict__ bb) {
  __shared__ float red[256];
  const int row = blockIdx.x, tid = threadIdx.x;
  const size_t base = (size_t)row * DMODEL;
  float v0 = x[base + tid]       + (a ? a[base + tid]       : 0.f);
  float v1 = x[base + tid + 256] + (a ? a[base + tid + 256] : 0.f);
  red[tid] = v0 + v1;
  __syncthreads();
  for (int s = 128; s > 0; s >>= 1) { if (tid < s) red[tid] += red[tid + s]; __syncthreads(); }
  float mean = red[0] * (1.f / DMODEL);
  __syncthreads();
  float d0 = v0 - mean, d1 = v1 - mean;
  red[tid] = d0 * d0 + d1 * d1;
  __syncthreads();
  for (int s = 128; s > 0; s >>= 1) { if (tid < s) red[tid] += red[tid + s]; __syncthreads(); }
  float inv = rsqrtf(red[0] * (1.f / DMODEL) + 1e-5f);
  float r0 = d0 * inv * g[tid]       + bb[tid];
  float r1 = d1 * inv * g[tid + 256] + bb[tid + 256];
  out[base + tid]       = r0;
  out[base + tid + 256] = r1;
  if (out16) {
    out16[base + tid]       = (_Float16)r0;
    out16[base + tid + 256] = (_Float16)r1;
  }
}

// ------------------------- final projection ---------------------------------
__global__ void __launch_bounds__(256)
proj_out(const float* __restrict__ hn, const float* __restrict__ pw,
         const float* __restrict__ pb, float* __restrict__ out) {
  int idx = blockIdx.x * 256 + threadIdx.x;
  if (idx >= BDIM * PRED * COUT) return;
  int cc = idx % COUT;
  int ti = (idx / COUT) % PRED;
  int b = idx / (COUT * PRED);
  int t = (T_LEN - PRED) + ti;
  const float* hr = hn + ((size_t)b * T_LEN + t) * DMODEL;
  float s = pb[cc];
  for (int e = 0; e < DMODEL; ++e) s += hr[e] * pw[(size_t)e * COUT + cc];
  out[idx] = s;
}

// -------------- weight convert fp32[K][N] -> fp16 transposed [N][K] ---------
__global__ void __launch_bounds__(256)
w_transpose_f16(const float* __restrict__ src, _Float16* __restrict__ dst,
                int K, int N) {
  int idx = blockIdx.x * 256 + threadIdx.x;
  if (idx >= K * N) return;
  int n = idx / K, k = idx - n * K;
  dst[idx] = (_Float16)src[(size_t)k * N + n];
}

// ---------------------------------------------------------------------------
extern "C" void kernel_launch(void* const* d_in, const int* in_sizes, int n_in,
                              void* d_out, int out_size, void* d_ws, size_t ws_size,
                              hipStream_t stream) {
  (void)in_sizes; (void)n_in; (void)out_size; (void)ws_size;
  const float* x_enc  = (const float*)d_in[0];
  const float* x_me   = (const float*)d_in[1];
  const float* x_dec  = (const float*)d_in[2];
  const float* x_md   = (const float*)d_in[3];
  const float* conv_w = (const float*)d_in[4];
  const float* mark_w = (const float*)d_in[5];
  const float* qk_w   = (const float*)d_in[6];
  const float* v_w    = (const float*)d_in[7];
  const float* out_w  = (const float*)d_in[8];
  const float* out_b  = (const float*)d_in[9];
  const float* ln1_g  = (const float*)d_in[10];
  const float* ln1_b  = (const float*)d_in[11];
  const float* ln2_g  = (const float*)d_in[12];
  const float* ln2_b  = (const float*)d_in[13];
  const float* ffn_w1 = (const float*)d_in[14];
  const float* ffn_b1 = (const float*)d_in[15];
  const float* ffn_w2 = (const float*)d_in[16];
  const float* ffn_b2 = (const float*)d_in[17];
  const float* lnf_g  = (const float*)d_in[18];
  const float* lnf_b  = (const float*)d_in[19];
  const float* proj_w = (const float*)d_in[20];
  const float* proj_b = (const float*)d_in[21];
  const float* rot    = (const float*)d_in[22];
  float* outp = (float*)d_out;
  char* ws = (char*)d_ws;

  constexpr size_t SZ_Wqk  = 2ull * DMODEL * DMODEL * 2;                    // 1 MB
  constexpr size_t SZ_Wff  = 2ull * DMODEL * DFF * 2;                       // 4 MB
  constexpr size_t SZ_ACT  = (size_t)BDIM * T_LEN * DMODEL * 4;             // 32 MB
  constexpr size_t SZ_ACTH = (size_t)BDIM * T_LEN * DMODEL * 2;             // 16 MB
  constexpr size_t SZ_OR   = (size_t)BH_TOT * NROUNDS * T_LEN * DHEAD * 2;  // 64 MB
  constexpr size_t SZ_LSE  = (size_t)BH_TOT * NROUNDS * T_LEN * 4;          // 2 MB

  size_t off = 0;
  const size_t oWqk = off; off += SZ_Wqk;
  const size_t oWv  = off; off += SZ_Wqk;
  const size_t oWo  = off; off += SZ_Wqk;
  const size_t oW1  = off; off += SZ_Wff;
  const size_t oW2  = off; off += SZ_Wff;
  const size_t oH   = off; off += SZ_ACT;
  const size_t oH16 = off; off += SZ_ACTH;
  const size_t oQK  = off; off += SZ_ACT;   // also merged f16 (16 MB)
  const size_t oV   = off; off += SZ_ACT;   // also ABUF f32 (32 MB)
  const size_t oOR  = off; off += SZ_OR;    // also FFN mid f16 (64 MB)
  const size_t oLSE = off; off += SZ_LSE;
  const size_t oST  = off; off += SZ_LSE;

  _Float16* Wqk16 = (_Float16*)(ws + oWqk);   // [l][N][K] transposed
  _Float16* Wv16  = (_Float16*)(ws + oWv);
  _Float16* Wo16  = (_Float16*)(ws + oWo);
  _Float16* W116  = (_Float16*)(ws + oW1);
  _Float16* W216  = (_Float16*)(ws + oW2);
  float*    H     = (float*)(ws + oH);
  _Float16* H16   = (_Float16*)(ws + oH16);
  float*    QK    = (float*)(ws + oQK);
  float*    Vb    = (float*)(ws + oV);
  _Float16* OR16  = (_Float16*)(ws + oOR);
  float*    LSE   = (float*)(ws + oLSE);
  int*      ST    = (int*)(ws + oST);
  _Float16* MERG  = (_Float16*)(ws + oQK);
  float*    ABUF  = (float*)(ws + oV);
  _Float16* FMID  = (_Float16*)(ws + oOR);

  // weights -> f16, transposed to [N][K] so GEMM B tiles are K-contiguous
  for (int l = 0; l < 2; ++l) {
    w_transpose_f16<<<1024, 256, 0, stream>>>(
        qk_w + (size_t)l * DMODEL * DMODEL, Wqk16 + (size_t)l * DMODEL * DMODEL, DMODEL, DMODEL);
    w_transpose_f16<<<1024, 256, 0, stream>>>(
        v_w + (size_t)l * DMODEL * DMODEL, Wv16 + (size_t)l * DMODEL * DMODEL, DMODEL, DMODEL);
    w_transpose_f16<<<1024, 256, 0, stream>>>(
        out_w + (size_t)l * DMODEL * DMODEL, Wo16 + (size_t)l * DMODEL * DMODEL, DMODEL, DMODEL);
    w_transpose_f16<<<4096, 256, 0, stream>>>(
        ffn_w1 + (size_t)l * DMODEL * DFF, W116 + (size_t)l * DMODEL * DFF, DMODEL, DFF);
    w_transpose_f16<<<4096, 256, 0, stream>>>(
        ffn_w2 + (size_t)l * DFF * DMODEL, W216 + (size_t)l * DFF * DMODEL, DFF, DMODEL);
  }

  embed_kernel<<<(BDIM * T_LEN * DMODEL) / 256, 256, 0, stream>>>(
      x_enc, x_me, x_dec, x_md, conv_w, mark_w, H, H16);

  const int M = BDIM * T_LEN;  // 16384
  const dim3 gD(M / BM, DMODEL / BN);   // (128, 8)
  const dim3 gF(M / BM, DFF / BN);      // (128, 32)
  const size_t SMEM = (2 * BM * BK + 2 * BN * BK) * sizeof(_Float16);  // 24 KB

  for (int l = 0; l < 2; ++l) {
    gemm_wmma<2, 0><<<gD, 256, SMEM, stream>>>(
        H16, Wqk16 + (size_t)l * DMODEL * DMODEL, nullptr, QK, M, DMODEL, DMODEL);
    gemm_wmma<2, 0><<<gD, 256, SMEM, stream>>>(
        H16, Wv16 + (size_t)l * DMODEL * DMODEL, nullptr, Vb, M, DMODEL, DMODEL);

    lsh_bucket_sort<<<dim3(BH_TOT, NROUNDS), 256, 0, stream>>>(
        QK, rot + (size_t)l * DHEAD * NROUNDS * 16, ST);

    lsh_attn<<<dim3(BH_TOT, NCHUNK), 256, 0, stream>>>(QK, Vb, ST, OR16, LSE);

    lsh_combine<<<(BH_TOT * T_LEN * DHEAD) / 256, 256, 0, stream>>>(OR16, LSE, MERG);

    gemm_wmma<0, 0><<<gD, 256, SMEM, stream>>>(
        MERG, Wo16 + (size_t)l * DMODEL * DMODEL, out_b + l * DMODEL, ABUF,
        M, DMODEL, DMODEL);
    add_ln<<<M, 256, 0, stream>>>(H, ABUF, H, H16,
                                  ln1_g + l * DMODEL, ln1_b + l * DMODEL);

    gemm_wmma<1, 1><<<gF, 256, SMEM, stream>>>(
        H16, W116 + (size_t)l * DMODEL * DFF, ffn_b1 + l * DFF, FMID, M, DFF, DMODEL);
    gemm_wmma<0, 0><<<gD, 256, SMEM, stream>>>(
        FMID, W216 + (size_t)l * DFF * DMODEL, ffn_b2 + l * DMODEL, ABUF,
        M, DMODEL, DFF);
    add_ln<<<M, 256, 0, stream>>>(H, ABUF, H, H16,
                                  ln2_g + l * DMODEL, ln2_b + l * DMODEL);
  }

  add_ln<<<M, 256, 0, stream>>>(H, nullptr, ABUF, nullptr, lnf_g, lnf_b);
  proj_out<<<(BDIM * PRED * COUT + 255) / 256, 256, 0, stream>>>(
      ABUF, proj_w, proj_b, outp);
}
